// Attention_64080912056468
// MI455X (gfx1250) — matrix operand order repaired and checked by reading, not compiled, not run on hardware
//
#include <hip/hip_runtime.h>

#define DIM_C   1024
#define HEADS_N 16
#define HD      64
#define SEQLEN  1024
#define BATCH   8
#define MROWS   (BATCH*SEQLEN)   // 8192
#define FQKV    (3*DIM_C)        // 3072

typedef __attribute__((ext_vector_type(16))) __bf16 v16bf;
typedef __attribute__((ext_vector_type(8)))  __bf16 v8bf;
typedef __attribute__((ext_vector_type(8)))  float  v8f;

union V16U { v16bf v; v8bf h[2]; };

__device__ __forceinline__ v8f wmma_bf16(v16bf a, v16bf b, v8f c) {
  // (neg_a, A, neg_b, B, c_mod, C, reuse_a, reuse_b)
  return __builtin_amdgcn_wmma_f32_16x16x32_bf16(false, a, false, b, (short)0, c, false, false);
}

// A-fragment (16x32 bf16): lane l16 = M row; per-lane K chunks [k0+8*hi,+8) and [k0+16+8*hi,+8)
__device__ __forceinline__ v16bf load_afrag(const __bf16* __restrict__ row, int k0, int hi) {
  V16U u;
  u.h[0] = *(const v8bf*)(row + k0 + 8*hi);
  u.h[1] = *(const v8bf*)(row + k0 + 16 + 8*hi);
  return u.v;
}

// ---------------- conversion / transpose ----------------
__global__ __launch_bounds__(256) void k_f32_to_bf16(const float* __restrict__ x,
                                                     __bf16* __restrict__ y, int n) {
  int i = blockIdx.x * 256 + threadIdx.x;
  if (i < n) y[i] = (__bf16)x[i];
}

// w: [R][C] f32 row-major  ->  wt: [C][R] bf16 (K-major for WMMA B operand)
__global__ __launch_bounds__(256) void k_transpose_bf16(const float* __restrict__ w,
                                                        __bf16* __restrict__ wt,
                                                        int R, int C) {
  int i = blockIdx.x * 256 + threadIdx.x;
  if (i < R * C) {
    int r = i / C, c = i % C;
    wt[(long)c * R + r] = (__bf16)w[i];
  }
}

// ---------------- QKV GEMM: qkv[m][f] = sum_c X[m][c] * Wqkv[f][c] ----------------
// Per-wave tile: 32 (M) x 64 (F): 2 A-fragments share 4 B-fragments -> 8 WMMA / k-step.
// X: [8192][1024] bf16, Wt: [1024][3072] bf16 (transposed Wqkv).
// Writes: Q [b][h][n][64], Kt [b][h][64][1024] (transposed), V [b][h][n][64], all bf16.
__global__ __launch_bounds__(128) void k_gemm_qkv(const __bf16* __restrict__ X,
                                                  const __bf16* __restrict__ Wt,
                                                  __bf16* __restrict__ Q,
                                                  __bf16* __restrict__ Kt,
                                                  __bf16* __restrict__ Vm) {
  const int w = threadIdx.x >> 5, lane = threadIdx.x & 31;
  const int l16 = lane & 15, hi = lane >> 4;
  const int mt = blockIdx.x & 63, nt = blockIdx.x >> 6;    // 64 x 48 blocks
  const int m0 = mt * 128 + w * 32, f0 = nt * 64;
  const __bf16* arow0 = X + (long)(m0 + l16) * DIM_C;
  const __bf16* arow1 = X + (long)(m0 + 16 + l16) * DIM_C;
  v8f acc[8] = {};
  for (int k0 = 0; k0 < DIM_C; k0 += 32) {
    v16bf a0 = load_afrag(arow0, k0, hi);
    v16bf a1 = load_afrag(arow1, k0, hi);
    const __bf16* brow = Wt + (long)(k0 + lane) * FQKV + f0;  // B lane = K row
    __builtin_prefetch(brow + 32 * FQKV, 0, 0);
#pragma unroll
    for (int f = 0; f < 4; ++f) {
      v16bf b;
#pragma unroll
      for (int s = 0; s < 16; ++s) { const int kk = (s < 8) ? 8 * hi + s : 16 + 8 * hi + (s - 8); b[s] = Wt[(long)(k0 + kk) * FQKV + f0 + f * 16 + l16]; }
      acc[f]     = wmma_bf16(a0, b, acc[f]);
      acc[4 + f] = wmma_bf16(a1, b, acc[4 + f]);
    }
  }
#pragma unroll
  for (int g = 0; g < 2; ++g) {
    const int mg = m0 + g * 16;
    const int bidx = mg >> 10, n0 = mg & 1023;
#pragma unroll
    for (int f = 0; f < 4; ++f) {
      const int col = f0 + f * 16 + l16;        // global output feature
      const int sel = col >> 10, c2 = col & 1023;
      const int h = c2 >> 6, d = c2 & 63;
      const long bh = (long)bidx * HEADS_N + h;
#pragma unroll
      for (int v = 0; v < 8; ++v) {
        const int n = n0 + v + 8 * hi;          // C layout: rows m = v + 8*hi
        const __bf16 val = (__bf16)acc[g * 4 + f][v];
        if (sel == 0)      Q [(bh * SEQLEN + n) * HD + d] = val;
        else if (sel == 1) Kt[(bh * HD + d) * SEQLEN + n] = val;
        else               Vm[(bh * SEQLEN + n) * HD + d] = val;
      }
    }
  }
}

// ---------------- axial RoPE ----------------
// pairs j=0..15 cover d=2j,2j+1; j<8 uses th[hp][j], j>=8 uses tw[wp][j-8]
__device__ __forceinline__ void rope_angle(int n, int j, float* c, float* s) {
  const int hp = n >> 5, wp = n & 31;
  const int pos = (j < 8) ? hp : wp;
  const int fi  = (j < 8) ? j  : j - 8;
  const float base = (1.0f + 73.0f * (float)fi) * 3.14159265358979323846f; // linspace(1,512,8)*pi
  const float ang  = (-1.0f + (2.0f / 31.0f) * (float)pos) * base;          // linspace(-1,1,32)
  *c = cosf(ang); *s = sinf(ang);
}

__global__ __launch_bounds__(256) void k_rope_q(__bf16* __restrict__ Q) {
  int i = blockIdx.x * 256 + threadIdx.x;   // BATCH*HEADS*SEQLEN*16 threads
  int j = i & 15; int t = i >> 4;
  int n = t & (SEQLEN - 1); int bh = t >> 10;
  float c, s; rope_angle(n, j, &c, &s);
  __bf16* p = Q + ((long)bh * SEQLEN + n) * HD + 2 * j;
  float x1 = (float)p[0], x2 = (float)p[1];
  p[0] = (__bf16)(x1 * c - x2 * s);
  p[1] = (__bf16)(x2 * c + x1 * s);
}

__global__ __launch_bounds__(256) void k_rope_kt(__bf16* __restrict__ Kt) {
  int i = blockIdx.x * 256 + threadIdx.x;
  int j = i & 15; int t = i >> 4;
  int n = t & (SEQLEN - 1); int bh = t >> 10;
  float c, s; rope_angle(n, j, &c, &s);
  __bf16* p0 = Kt + ((long)bh * HD + 2 * j) * SEQLEN + n;
  __bf16* p1 = p0 + SEQLEN;
  float x1 = (float)*p0, x2 = (float)*p1;
  *p0 = (__bf16)(x1 * c - x2 * s);
  *p1 = (__bf16)(x2 * c + x1 * s);
}

// ---------------- flash attention ----------------
// Online-softmax update for one 16-row group over a 32-key step.
__device__ __forceinline__ void softmax_update(v8f sA, v8f sB,
                                               float* __restrict__ rmax,
                                               float* __restrict__ rsum,
                                               float* __restrict__ pa,
                                               float* __restrict__ pb,
                                               v8f* __restrict__ o4) {
#pragma unroll
  for (int v = 0; v < 8; ++v) {
    float xs = sA[v] * 0.125f, ys = sB[v] * 0.125f;     // 1/sqrt(64)
    float m = fmaxf(xs, ys);                             // row max across 16 lanes (half-wave)
    m = fmaxf(m, __shfl_xor(m, 1));
    m = fmaxf(m, __shfl_xor(m, 2));
    m = fmaxf(m, __shfl_xor(m, 4));
    m = fmaxf(m, __shfl_xor(m, 8));
    float nm   = fmaxf(rmax[v], m);
    float corr = __expf(rmax[v] - nm);
    rmax[v] = nm;
    pa[v] = __expf(xs - nm);
    pb[v] = __expf(ys - nm);
    float ts = pa[v] + pb[v];
    ts += __shfl_xor(ts, 1);
    ts += __shfl_xor(ts, 2);
    ts += __shfl_xor(ts, 4);
    ts += __shfl_xor(ts, 8);
    rsum[v] = rsum[v] * corr + ts;
    o4[0][v] *= corr; o4[1][v] *= corr;
    o4[2][v] *= corr; o4[3][v] *= corr;
  }
}

// block = 256 thr (8 waves); wave handles 32 q rows (two 16-row groups sharing K/V
// fragments); grid = B*H*(1024/256). 16 WMMAs per 32-key step per wave.
__global__ __launch_bounds__(256) void k_attention(const __bf16* __restrict__ Q,
                                                   const __bf16* __restrict__ Kt,
                                                   const __bf16* __restrict__ Vm,
                                                   __bf16* __restrict__ O) {
  __shared__ __bf16 Plds[8][2][16][32] __attribute__((aligned(32)));
  const int w = threadIdx.x >> 5, lane = threadIdx.x & 31;
  const int l16 = lane & 15, hi = lane >> 4;
  const int bh = blockIdx.x >> 2;
  const int q0 = (blockIdx.x & 3) * 256 + w * 32;
  const __bf16* qb  = Q  + (long)bh * SEQLEN * HD;
  const __bf16* ktb = Kt + (long)bh * HD * SEQLEN;
  const __bf16* vb  = Vm + (long)bh * SEQLEN * HD;
  const __bf16* qrow0 = qb + (long)(q0 + l16) * HD;
  const __bf16* qrow1 = qb + (long)(q0 + 16 + l16) * HD;
  const v16bf a0 = load_afrag(qrow0, 0, hi), a1 = load_afrag(qrow0, 32, hi);
  const v16bf a2 = load_afrag(qrow1, 0, hi), a3 = load_afrag(qrow1, 32, hi);
  v8f oacc[8] = {};
  float rmax[16], rsum[16];
#pragma unroll
  for (int v = 0; v < 16; ++v) { rmax[v] = -1.0e30f; rsum[v] = 0.0f; }
  const __bf16* kr0 = ktb + (long)lane * SEQLEN;          // kd rows 0..31  (B lane = K)
  const __bf16* kr1 = ktb + (long)(lane + 32) * SEQLEN;   // kd rows 32..63

  for (int kb = 0; kb < SEQLEN; kb += 32) {
    v16bf b0, b1, b2, b3;
#pragma unroll
    for (int s = 0; s < 16; ++s) {
      const int kk = (s < 8) ? 8 * hi + s : 16 + 8 * hi + (s - 8);
      b0[s] = ktb[(long)kk * SEQLEN + kb + l16];
      b1[s] = ktb[(long)(kk + 32) * SEQLEN + kb + l16];
      b2[s] = ktb[(long)kk * SEQLEN + kb + 16 + l16];
      b3[s] = ktb[(long)(kk + 32) * SEQLEN + kb + 16 + l16];
    }
    v8f sA0 = {}, sB0 = {}, sA1 = {}, sB1 = {};
    sA0 = wmma_bf16(a0, b0, sA0);  sA0 = wmma_bf16(a1, b1, sA0);
    sB0 = wmma_bf16(a0, b2, sB0);  sB0 = wmma_bf16(a1, b3, sB0);
    sA1 = wmma_bf16(a2, b0, sA1);  sA1 = wmma_bf16(a3, b1, sA1);
    sB1 = wmma_bf16(a2, b2, sB1);  sB1 = wmma_bf16(a3, b3, sB1);

    float pa0[8], pb0[8], pa1[8], pb1[8];
    softmax_update(sA0, sB0, rmax,     rsum,     pa0, pb0, &oacc[0]);
    softmax_update(sA1, sB1, rmax + 8, rsum + 8, pa1, pb1, &oacc[4]);

    // C-layout (lane=key) -> A-layout (lane=row) transpose via LDS
#pragma unroll
    for (int v = 0; v < 8; ++v) {
      Plds[w][0][v + 8 * hi][l16]      = (__bf16)pa0[v];
      Plds[w][0][v + 8 * hi][16 + l16] = (__bf16)pb0[v];
      Plds[w][1][v + 8 * hi][l16]      = (__bf16)pa1[v];
      Plds[w][1][v + 8 * hi][16 + l16] = (__bf16)pb1[v];
    }
    asm volatile("s_wait_dscnt 0x0" ::: "memory");
    V16U pf0, pf1;
    pf0.h[0] = *(const v8bf*)&Plds[w][0][l16][8 * hi];
    pf0.h[1] = *(const v8bf*)&Plds[w][0][l16][16 + 8 * hi];
    pf1.h[0] = *(const v8bf*)&Plds[w][1][l16][8 * hi];
    pf1.h[1] = *(const v8bf*)&Plds[w][1][l16][16 + 8 * hi];

    const __bf16* vrow = vb + (long)(kb + lane) * HD;     // B lane = key row
#pragma unroll
    for (int f = 0; f < 4; ++f) {
      v16bf bv;
#pragma unroll
      for (int s = 0; s < 16; ++s) { const int kk = (s < 8) ? 8 * hi + s : 16 + 8 * hi + (s - 8); bv[s] = vb[(long)(kb + kk) * HD + f * 16 + l16]; }
      oacc[f]     = wmma_bf16(pf0.v, bv, oacc[f]);
      oacc[4 + f] = wmma_bf16(pf1.v, bv, oacc[4 + f]);
    }
  }
  const int b = bh >> 4, h = bh & 15;
#pragma unroll
  for (int g = 0; g < 2; ++g) {
#pragma unroll
    for (int f = 0; f < 4; ++f) {
      const int d = f * 16 + l16;
#pragma unroll
      for (int v = 0; v < 8; ++v) {
        const int n = q0 + g * 16 + v + 8 * hi;
        O[((long)(b * SEQLEN + n)) * DIM_C + h * HD + d] =
            (__bf16)(oacc[g * 4 + f][v] / rsum[g * 8 + v]);
      }
    }
  }
}

// ---------------- output projection (fp32 out + bias) ----------------
__global__ __launch_bounds__(128) void k_gemm_proj(const __bf16* __restrict__ X,
                                                   const __bf16* __restrict__ Wt,
                                                   const float* __restrict__ bias,
                                                   float* __restrict__ out) {
  const int w = threadIdx.x >> 5, lane = threadIdx.x & 31;
  const int l16 = lane & 15, hi = lane >> 4;
  const int mt = blockIdx.x & 63, nt = blockIdx.x >> 6;    // 64 x 16 blocks
  const int m0 = mt * 128 + w * 32, f0 = nt * 64;
  const __bf16* arow0 = X + (long)(m0 + l16) * DIM_C;
  const __bf16* arow1 = X + (long)(m0 + 16 + l16) * DIM_C;
  v8f acc[8] = {};
  for (int k0 = 0; k0 < DIM_C; k0 += 32) {
    v16bf a0 = load_afrag(arow0, k0, hi);
    v16bf a1 = load_afrag(arow1, k0, hi);
    const __bf16* brow = Wt + (long)(k0 + lane) * DIM_C + f0;
    __builtin_prefetch(brow + 32 * DIM_C, 0, 0);
#pragma unroll
    for (int f = 0; f < 4; ++f) {
      v16bf b;
#pragma unroll
      for (int s = 0; s < 16; ++s) { const int kk = (s < 8) ? 8 * hi + s : 16 + 8 * hi + (s - 8); b[s] = Wt[(long)(k0 + kk) * DIM_C + f0 + f * 16 + l16]; }
      acc[f]     = wmma_bf16(a0, b, acc[f]);
      acc[4 + f] = wmma_bf16(a1, b, acc[4 + f]);
    }
  }
#pragma unroll
  for (int g = 0; g < 2; ++g) {
#pragma unroll
    for (int f = 0; f < 4; ++f) {
      const int col = f0 + f * 16 + l16;
      const float bv = bias[col];
#pragma unroll
      for (int v = 0; v < 8; ++v) {
        const int m = m0 + g * 16 + v + 8 * hi;
        out[(long)m * DIM_C + col] = acc[g * 4 + f][v] + bv;
      }
    }
  }
}

extern "C" void kernel_launch(void* const* d_in, const int* in_sizes, int n_in,
                              void* d_out, int out_size, void* d_ws, size_t ws_size,
                              hipStream_t stream) {
  (void)in_sizes; (void)n_in; (void)out_size; (void)ws_size;
  const float* x     = (const float*)d_in[0];   // [8,1024,1024]
  const float* wqkv  = (const float*)d_in[1];   // [3072,1024]
  const float* wproj = (const float*)d_in[2];   // [1024,1024]
  const float* bproj = (const float*)d_in[3];   // [1024]
  float* out = (float*)d_out;

  char* ws = (char*)d_ws;
  __bf16* xb     = (__bf16*)(ws);                          // 16 MiB
  __bf16* wqkvt  = (__bf16*)(ws + (16ull  << 20));         //  6 MiB  [1024][3072]
  __bf16* wprojt = (__bf16*)(ws + (22ull  << 20));         //  2 MiB  [1024][1024]
  __bf16* qbuf   = (__bf16*)(ws + (24ull  << 20));         // 16 MiB  [b][h][n][64]
  __bf16* ktbuf  = (__bf16*)(ws + (40ull  << 20));         // 16 MiB  [b][h][64][n]
  __bf16* vbuf   = (__bf16*)(ws + (56ull  << 20));         // 16 MiB  [b][h][n][64]
  __bf16* attnb  = (__bf16*)(ws + (72ull  << 20));         // 16 MiB  [b*n][1024]

  const int nx = MROWS * DIM_C;                 // 8388608
  k_f32_to_bf16<<<(nx + 255) / 256, 256, 0, stream>>>(x, xb, nx);
  k_transpose_bf16<<<(FQKV * DIM_C + 255) / 256, 256, 0, stream>>>(wqkv,  wqkvt,  FQKV,  DIM_C);
  k_transpose_bf16<<<(DIM_C * DIM_C + 255) / 256, 256, 0, stream>>>(wproj, wprojt, DIM_C, DIM_C);

  k_gemm_qkv<<<(MROWS / 128) * (FQKV / 64), 128, 0, stream>>>(xb, wqkvt, qbuf, ktbuf, vbuf);

  const int nrope = BATCH * HEADS_N * SEQLEN * 16;  // 2097152
  k_rope_q <<<nrope / 256, 256, 0, stream>>>(qbuf);
  k_rope_kt<<<nrope / 256, 256, 0, stream>>>(ktbuf);

  k_attention<<<BATCH * HEADS_N * (SEQLEN / 256), 256, 0, stream>>>(qbuf, ktbuf, vbuf, attnb);

  k_gemm_proj<<<(MROWS / 128) * (DIM_C / 64), 128, 0, stream>>>(attnb, wprojt, bproj, out);
}
